// Text_Mmamba_23905787969608
// MI455X (gfx1250) — matrix operand order repaired and checked by reading, not compiled, not run on hardware
//
#include <hip/hip_runtime.h>
#include <hip/hip_bf16.h>
#include <math.h>

// ---------------- model dims ----------------
#define D_MODEL   1024
#define D_STATE   128
#define D_CONV    4
#define HEADDIM   64
#define D_INNER   2048
#define NHEADS    32
#define CONV_DIM  2304            // D_INNER + 2*D_STATE
#define D_IN_PROJ 4384            // 2*D_INNER + 2*D_STATE + NHEADS
#define N_LAYERS  12
#define BATCH     2
#define TCODE     512
#define CONDL     64
#define SEQ       576             // CONDL + TCODE
#define OUTL      513             // SEQ - (CONDL-1)
#define VOCAB     1024
#define CODEC     4
#define COND_DIM  768
#define ROWS      (BATCH*SEQ)     // 1152
#define EPSV      1e-5f

typedef __bf16 bf16;
typedef __attribute__((ext_vector_type(16))) __bf16 v16bf;
typedef __attribute__((ext_vector_type(8)))  float  v8f;
typedef __attribute__((ext_vector_type(4)))  float  f32x4;
typedef __attribute__((ext_vector_type(4)))  __bf16 bf16x4;

enum { EP_STORE = 0, EP_BIAS = 1, EP_BIAS_GELU = 2, EP_RES = 3, EP_BIAS_RES = 4, EP_LMHEAD = 5 };

// ------------------------------------------------------------------
// GEMM:  C[M,N] = epilogue( A[M,K] (f32) * W[N,K]^T (f32) )
// bf16 WMMA 16x16x32, fp32 accumulate.
// Block = 256 threads = 8 waves, 128x64 C tile, each wave a 32x32
// subtile (2x2 WMMAs). Double-buffered LDS, vectorized branch-free
// staging (global b128 loads -> packed cvt -> ds b64 stores).
// Requires: M % 128 == 0, K % 32 == 0 (true for all call sites).
// ------------------------------------------------------------------
#define LDST 40   // padded bf16 row stride (16B-aligned rows, bank-conflict-free)

__global__ __launch_bounds__(256)
void k_gemm(const float* __restrict__ A, const float* __restrict__ W,
            float* __restrict__ out, const float* __restrict__ bias,
            const float* __restrict__ res, int M, int N, int K, int mode)
{
    __shared__ bf16 la[2][128 * LDST];
    __shared__ bf16 lw[2][64 * LDST];
    const int tid  = threadIdx.x;
    const int lane = tid & 31;
    const int wave = tid >> 5;
    const int m0 = blockIdx.y * 128;
    const int n0 = blockIdx.x * 64;
    const int wm = (wave >> 1) * 32;       // 0,32,64,96
    const int wn = (wave & 1) * 32;        // 0,32
    const bool wfull = (n0 + 64 <= N);     // uniform: only false for in_proj edge block

    f32x4 ra[4], rw[2];

    // A tile: 128 rows x 32 cols = 1024 float4 -> 4 per thread
    // W tile:  64 rows x 32 cols =  512 float4 -> 2 per thread
    auto load_tiles = [&](int k0) {
#pragma unroll
        for (int i = 0; i < 4; ++i) {
            int id = i * 256 + tid;
            int r = id >> 3, c4 = id & 7;
            ra[i] = *(const f32x4*)(A + (size_t)(m0 + r) * K + k0 + c4 * 4);
        }
        if (wfull) {
#pragma unroll
            for (int i = 0; i < 2; ++i) {
                int id = i * 256 + tid;
                int r = id >> 3, c4 = id & 7;
                rw[i] = *(const f32x4*)(W + (size_t)(n0 + r) * K + k0 + c4 * 4);
            }
        } else {
#pragma unroll
            for (int i = 0; i < 2; ++i) {
                int id = i * 256 + tid;
                int r = id >> 3, c4 = id & 7;
                if (n0 + r < N)
                    rw[i] = *(const f32x4*)(W + (size_t)(n0 + r) * K + k0 + c4 * 4);
                else { rw[i][0] = 0.f; rw[i][1] = 0.f; rw[i][2] = 0.f; rw[i][3] = 0.f; }
            }
        }
    };

    auto store_tiles = [&](int buf) {
#pragma unroll
        for (int i = 0; i < 4; ++i) {
            int id = i * 256 + tid;
            int r = id >> 3, c4 = id & 7;
            bf16x4 v;
#pragma unroll
            for (int e = 0; e < 4; ++e) v[e] = (bf16)ra[i][e];
            *(bf16x4*)&la[buf][r * LDST + c4 * 4] = v;
        }
#pragma unroll
        for (int i = 0; i < 2; ++i) {
            int id = i * 256 + tid;
            int r = id >> 3, c4 = id & 7;
            bf16x4 v;
#pragma unroll
            for (int e = 0; e < 4; ++e) v[e] = (bf16)rw[i][e];
            *(bf16x4*)&lw[buf][r * LDST + c4 * 4] = v;
        }
    };

    v8f acc[2][2];
#pragma unroll
    for (int i = 0; i < 2; ++i)
#pragma unroll
        for (int j = 0; j < 2; ++j)
#pragma unroll
            for (int r = 0; r < 8; ++r) acc[i][j][r] = 0.f;

    load_tiles(0);
    int buf = 0;
    for (int k0 = 0; k0 < K; k0 += 32) {
        store_tiles(buf);
        __syncthreads();
        if (k0 + 32 < K) load_tiles(k0 + 32);   // prefetch next tile while computing

        // Fragments per ISA 7.12.2 (wave32):
        // A 16x32: lane<16 -> row M=lane, K {0..7,16..23}; lane>=16 -> row M=lane-16, K {8..15,24..31}
        v16bf af[2], bfr[2];
        {
            int kb = (lane < 16) ? 0 : 8;
#pragma unroll
            for (int ti = 0; ti < 2; ++ti) {
                const bf16* p = &la[buf][(wm + ti * 16 + (lane & 15)) * LDST];
#pragma unroll
                for (int e = 0; e < 8; ++e) af[ti][e] = p[kb + e];
#pragma unroll
                for (int e = 0; e < 8; ++e) af[ti][8 + e] = p[16 + kb + e];
            }
            // B 32x16 (same K order as A): lanes<16 -> col N=lane, K {0..7,16..23}; lanes>=16 -> K {8..15,24..31}
            int kb2 = (lane < 16) ? 0 : 8;
#pragma unroll
            for (int tj = 0; tj < 2; ++tj) {
                const bf16* p = &lw[buf][(wn + tj * 16 + (lane & 15)) * LDST];
#pragma unroll
                for (int e = 0; e < 8; ++e) bfr[tj][e] = p[kb2 + e];
#pragma unroll
                for (int e = 0; e < 8; ++e) bfr[tj][8 + e] = p[16 + kb2 + e];
            }
        }
#pragma unroll
        for (int ti = 0; ti < 2; ++ti)
#pragma unroll
            for (int tj = 0; tj < 2; ++tj)
                acc[ti][tj] = __builtin_amdgcn_wmma_f32_16x16x32_bf16(
                    false, af[ti], false, bfr[tj], (short)0, acc[ti][tj], false, false);
        buf ^= 1;
    }

    // epilogue: C/D layout -> VGPR r, lane: M = r + (lane>=16)*8, N = lane&15
#pragma unroll
    for (int ti = 0; ti < 2; ++ti) {
#pragma unroll
        for (int tj = 0; tj < 2; ++tj) {
            int n = n0 + wn + tj * 16 + (lane & 15);
            if (n >= N) continue;
#pragma unroll
            for (int r = 0; r < 8; ++r) {
                int m = m0 + wm + ti * 16 + r + ((lane >= 16) ? 8 : 0);
                float v = acc[ti][tj][r];
                if (mode == EP_BIAS) v += bias[n];
                else if (mode == EP_BIAS_GELU) {
                    v += bias[n];
                    float x3 = v * v * v;
                    v = 0.5f * v * (1.f + tanhf(0.7978845608028654f * (v + 0.044715f * x3)));
                } else if (mode == EP_RES) v += res[(size_t)m * N + n];
                else if (mode == EP_BIAS_RES) v += bias[n] + res[(size_t)m * N + n];
                if (mode == EP_LMHEAD) {
                    int b = m / SEQ, l = m % SEQ;
                    if (l >= CONDL - 1)
                        out[(((size_t)(b * CODEC + (n >> 10))) * OUTL + (l - (CONDL - 1))) * VOCAB + (n & 1023)] = v;
                } else {
                    out[(size_t)m * N + n] = v;
                }
            }
        }
    }
}

// ---------------- block reduction helper (blockDim == 256) ----------------
__device__ __forceinline__ float block_sum(float v, float* sm)
{
    int t = threadIdx.x;
    sm[t] = v;
    __syncthreads();
#pragma unroll
    for (int s = 128; s > 0; s >>= 1) {
        if (t < s) sm[t] += sm[t + s];
        __syncthreads();
    }
    float r = sm[0];
    __syncthreads();
    return r;
}

// ---------------- embedding sum (4 codebooks) ----------------
__global__ __launch_bounds__(256) void k_embed(const int* __restrict__ x,
                                               const float* __restrict__ emb,
                                               float* __restrict__ h)
{
    int idx = blockIdx.x * blockDim.x + threadIdx.x;      // B*TCODE*D_MODEL
    if (idx >= BATCH * TCODE * D_MODEL) return;
    int d = idx & (D_MODEL - 1);
    int l = (idx >> 10) % TCODE;
    int b = idx / (TCODE * D_MODEL);
    float s = 0.f;
#pragma unroll
    for (int c = 0; c < CODEC; ++c) {
        int tok = x[(b * CODEC + c) * TCODE + l];
        s += emb[((size_t)(c * VOCAB + tok)) * D_MODEL + d];
    }
    h[((size_t)(b * SEQ + CONDL + l)) * D_MODEL + d] = s;
}

// ---------------- sinusoidal positional embedding add ----------------
__global__ __launch_bounds__(256) void k_sinadd(float* __restrict__ h)
{
    int idx = blockIdx.x * blockDim.x + threadIdx.x;      // B*SEQ*D_MODEL
    if (idx >= BATCH * SEQ * D_MODEL) return;
    int d = idx & (D_MODEL - 1);
    int l = (idx >> 10) % SEQ;
    int a = (d < 512) ? d : d - 512;
    float freq = powf(10000.f, -(float)a / 511.f);
    float ph = (float)l * freq;
    h[idx] += (d < 512) ? cosf(ph) : sinf(ph);
}

// ---------------- layernorm of cond rows, scattered into h ----------------
__global__ __launch_bounds__(256) void k_ln_cond(const float* __restrict__ in,
                                                 const float* __restrict__ w,
                                                 const float* __restrict__ bias,
                                                 float* __restrict__ h)
{
    __shared__ float sm[256];
    int row = blockIdx.x;                                 // 0..127 (b*64+l)
    int t = threadIdx.x;
    const float* p = in + (size_t)row * D_MODEL;
    float s = 0.f;
    for (int d = t; d < D_MODEL; d += 256) s += p[d];
    float mu = block_sum(s, sm) / (float)D_MODEL;
    float s2 = 0.f;
    for (int d = t; d < D_MODEL; d += 256) { float dv = p[d] - mu; s2 += dv * dv; }
    float var = block_sum(s2, sm) / (float)D_MODEL;
    float inv = rsqrtf(var + EPSV);
    int b = row / CONDL, l = row % CONDL;
    float* o = h + ((size_t)(b * SEQ + l)) * D_MODEL;
    for (int d = t; d < D_MODEL; d += 256) o[d] = (p[d] - mu) * inv * w[d] + bias[d];
}

// ---------------- rmsnorm ----------------
__global__ __launch_bounds__(256) void k_rmsnorm(const float* __restrict__ in,
                                                 const float* __restrict__ w,
                                                 float* __restrict__ out, int dim)
{
    __shared__ float sm[256];
    int row = blockIdx.x;
    int t = threadIdx.x;
    const float* p = in + (size_t)row * dim;
    float s = 0.f;
    for (int d = t; d < dim; d += 256) { float v = p[d]; s += v * v; }
    float inv = rsqrtf(block_sum(s, sm) / (float)dim + EPSV);
    float* o = out + (size_t)row * dim;
    for (int d = t; d < dim; d += 256) o[d] = p[d] * inv * w[d];
}

// ---------------- causal depthwise conv(4) + SiLU ----------------
__global__ __launch_bounds__(256) void k_conv(const float* __restrict__ zx,
                                              const float* __restrict__ cw,
                                              const float* __restrict__ cb,
                                              float* __restrict__ xBCc)
{
    int id = blockIdx.x * blockDim.x + threadIdx.x;
    if (id >= BATCH * CONV_DIM) return;
    int b = id / CONV_DIM, c = id % CONV_DIM;
    float w0 = cw[c * D_CONV + 0], w1 = cw[c * D_CONV + 1];
    float w2 = cw[c * D_CONV + 2], w3 = cw[c * D_CONV + 3];
    float bb = cb[c];
    float x0 = 0.f, x1 = 0.f, x2 = 0.f, x3 = 0.f;
    const float* src = zx + (size_t)(b * SEQ) * D_IN_PROJ + D_INNER + c;
    float* dst = xBCc + (size_t)(b * SEQ) * CONV_DIM + c;
    for (int l = 0; l < SEQ; ++l) {
        x0 = x1; x1 = x2; x2 = x3;
        x3 = src[(size_t)l * D_IN_PROJ];
        float v = bb + w0 * x0 + w1 * x1 + w2 * x2 + w3 * x3;
        v = v / (1.f + expf(-v));                         // SiLU
        dst[(size_t)l * CONV_DIM] = v;
    }
}

// ---------------- dt = softplus(raw + dt_bias) ----------------
__global__ __launch_bounds__(256) void k_dt(const float* __restrict__ zx,
                                            const float* __restrict__ dbias,
                                            float* __restrict__ dt)
{
    int id = blockIdx.x * blockDim.x + threadIdx.x;       // ROWS*NHEADS
    if (id >= ROWS * NHEADS) return;
    int m = id >> 5, hh = id & 31;
    float v = zx[(size_t)m * D_IN_PROJ + (D_IN_PROJ - NHEADS) + hh] + dbias[hh];
    dt[id] = (v > 20.f) ? v : log1pf(expf(v));
}

// ---------------- selective scan: one block per (batch, head) ----------------
// state (64x128) kept in registers: 32 f32 per thread; B/C/x staged via LDS.
__global__ __launch_bounds__(256) void k_scan(const float* __restrict__ xBCc,
                                              const float* __restrict__ dt,
                                              const float* __restrict__ A_log,
                                              const float* __restrict__ Dp,
                                              float* __restrict__ y)
{
    int b  = blockIdx.x >> 5;
    int hh = blockIdx.x & 31;
    int t  = threadIdx.x;
    int p  = t >> 2;                 // 0..63 (headdim)
    int n0 = (t & 3) * 32;           // 32-wide n-chunk
    float st[32];
#pragma unroll
    for (int i = 0; i < 32; ++i) st[i] = 0.f;
    float Ah = -expf(A_log[hh]);
    float Dv = Dp[hh];
    __shared__ float xs[HEADDIM];
    __shared__ float Bs[D_STATE];
    __shared__ float Cs[D_STATE];
    for (int l = 0; l < SEQ; ++l) {
        const float* row = xBCc + (size_t)(b * SEQ + l) * CONV_DIM;
        __syncthreads();
        if (t < 128) Bs[t] = row[D_INNER + t];
        else         Cs[t - 128] = row[D_INNER + D_STATE + (t - 128)];
        if (t < 64)  xs[t] = row[hh * HEADDIM + t];
        __syncthreads();
        float dtl = dt[(size_t)(b * SEQ + l) * NHEADS + hh];
        float dA  = expf(dtl * Ah);
        float xv  = xs[p];
        float cm  = dtl * xv;
        float acc = 0.f;
#pragma unroll
        for (int i = 0; i < 32; ++i) {
            float s = st[i] * dA + cm * Bs[n0 + i];
            st[i] = s;
            acc += s * Cs[n0 + i];
        }
        acc += __shfl_xor(acc, 1, 32);
        acc += __shfl_xor(acc, 2, 32);
        if ((t & 3) == 0)
            y[(size_t)(b * SEQ + l) * D_INNER + hh * HEADDIM + p] = acc + Dv * xv;
    }
}

// ---------------- gated rmsnorm: rmsnorm(y * silu(z)) * w ----------------
__global__ __launch_bounds__(256) void k_gated(const float* __restrict__ y,
                                               const float* __restrict__ zx,
                                               const float* __restrict__ w,
                                               float* __restrict__ out)
{
    __shared__ float sm[256];
    int m = blockIdx.x;
    int t = threadIdx.x;
    float vals[D_INNER / 256];
    float s = 0.f;
#pragma unroll
    for (int i = 0; i < D_INNER / 256; ++i) {
        int d = t + i * 256;
        float z = zx[(size_t)m * D_IN_PROJ + d];
        float g = z / (1.f + expf(-z));
        float v = y[(size_t)m * D_INNER + d] * g;
        vals[i] = v; s += v * v;
    }
    float inv = rsqrtf(block_sum(s, sm) / (float)D_INNER + EPSV);
#pragma unroll
    for (int i = 0; i < D_INNER / 256; ++i) {
        int d = t + i * 256;
        out[(size_t)m * D_INNER + d] = vals[i] * inv * w[d];
    }
}

// ------------------------------------------------------------------
extern "C" void kernel_launch(void* const* d_in, const int* in_sizes, int n_in,
                              void* d_out, int out_size, void* d_ws, size_t ws_size,
                              hipStream_t stream)
{
    (void)in_sizes; (void)n_in; (void)out_size; (void)ws_size;
    const int*   x          = (const int*)  d_in[0];
    const float* condition  = (const float*)d_in[1];
    /* d_in[2] = condition_mask, unused by reference */
    const float* emb        = (const float*)d_in[3];
    const float* cond_lin_w = (const float*)d_in[4];
    const float* cond_lin_b = (const float*)d_in[5];
    const float* cond_ln_w  = (const float*)d_in[6];
    const float* cond_ln_b  = (const float*)d_in[7];
    const float* m_norm_w   = (const float*)d_in[8];
    const float* in_proj_w  = (const float*)d_in[9];
    const float* conv_w     = (const float*)d_in[10];
    const float* conv_b     = (const float*)d_in[11];
    const float* A_log      = (const float*)d_in[12];
    const float* dt_bias    = (const float*)d_in[13];
    const float* Dparam     = (const float*)d_in[14];
    const float* ssm_norm_w = (const float*)d_in[15];
    const float* out_proj_w = (const float*)d_in[16];
    const float* f_norm_w   = (const float*)d_in[17];
    const float* w1         = (const float*)d_in[18];
    const float* b1         = (const float*)d_in[19];
    const float* w2         = (const float*)d_in[20];
    const float* b2         = (const float*)d_in[21];
    const float* lm_head    = (const float*)d_in[22];

    float* ws   = (float*)d_ws;
    float* h    = ws; ws += (size_t)ROWS * D_MODEL;
    float* hn   = ws; ws += (size_t)ROWS * D_MODEL;
    float* zx   = ws; ws += (size_t)ROWS * D_IN_PROJ;
    float* xBCc = ws; ws += (size_t)ROWS * CONV_DIM;
    float* dtb  = ws; ws += (size_t)ROWS * NHEADS;
    float* ybuf = ws; ws += (size_t)ROWS * D_INNER;
    float* ynb  = ws; ws += (size_t)ROWS * D_INNER;
    float* mlp1 = ws; ws += (size_t)ROWS * D_INNER;
    float* ctmp = ws; ws += (size_t)(BATCH * CONDL) * D_MODEL;

    const dim3 gB(256);

    // ---- stem: cond linear + layernorm, embeddings, sin PE ----
    k_gemm<<<dim3(D_MODEL / 64, (BATCH * CONDL) / 128), gB, 0, stream>>>(
        condition, cond_lin_w, ctmp, cond_lin_b, nullptr,
        BATCH * CONDL, D_MODEL, COND_DIM, EP_BIAS);
    k_ln_cond<<<BATCH * CONDL, 256, 0, stream>>>(ctmp, cond_ln_w, cond_ln_b, h);
    k_embed<<<(BATCH * TCODE * D_MODEL) / 256, 256, 0, stream>>>(x, emb, h);
    k_sinadd<<<(BATCH * SEQ * D_MODEL) / 256, 256, 0, stream>>>(h);

    // ---- 12 Mamba2 layers ----
    for (int L = 0; L < N_LAYERS; ++L) {
        const float* mnw = m_norm_w   + (size_t)L * D_MODEL;
        const float* ipw = in_proj_w  + (size_t)L * D_IN_PROJ * D_MODEL;
        const float* cw  = conv_w     + (size_t)L * CONV_DIM * D_CONV;
        const float* cb  = conv_b     + (size_t)L * CONV_DIM;
        const float* al  = A_log      + (size_t)L * NHEADS;
        const float* dbi = dt_bias    + (size_t)L * NHEADS;
        const float* Dp  = Dparam     + (size_t)L * NHEADS;
        const float* snw = ssm_norm_w + (size_t)L * D_INNER;
        const float* opw = out_proj_w + (size_t)L * D_MODEL * D_INNER;
        const float* fnw = f_norm_w   + (size_t)L * D_MODEL;
        const float* w1p = w1 + (size_t)L * D_INNER * D_MODEL;
        const float* b1p = b1 + (size_t)L * D_INNER;
        const float* w2p = w2 + (size_t)L * D_MODEL * D_INNER;
        const float* b2p = b2 + (size_t)L * D_MODEL;

        k_rmsnorm<<<ROWS, 256, 0, stream>>>(h, mnw, hn, D_MODEL);
        k_gemm<<<dim3((D_IN_PROJ + 63) / 64, ROWS / 128), gB, 0, stream>>>(
            hn, ipw, zx, nullptr, nullptr, ROWS, D_IN_PROJ, D_MODEL, EP_STORE);
        k_conv<<<(BATCH * CONV_DIM + 255) / 256, 256, 0, stream>>>(zx, cw, cb, xBCc);
        k_dt<<<(ROWS * NHEADS) / 256, 256, 0, stream>>>(zx, dbi, dtb);
        k_scan<<<BATCH * NHEADS, 256, 0, stream>>>(xBCc, dtb, al, Dp, ybuf);
        k_gated<<<ROWS, 256, 0, stream>>>(ybuf, zx, snw, ynb);
        k_gemm<<<dim3(D_MODEL / 64, ROWS / 128), gB, 0, stream>>>(
            ynb, opw, h, nullptr, h, ROWS, D_MODEL, D_INNER, EP_RES);
        k_rmsnorm<<<ROWS, 256, 0, stream>>>(h, fnw, hn, D_MODEL);
        k_gemm<<<dim3(D_INNER / 64, ROWS / 128), gB, 0, stream>>>(
            hn, w1p, mlp1, b1p, nullptr, ROWS, D_INNER, D_MODEL, EP_BIAS_GELU);
        k_gemm<<<dim3(D_MODEL / 64, ROWS / 128), gB, 0, stream>>>(
            mlp1, w2p, h, b2p, h, ROWS, D_MODEL, D_INNER, EP_BIAS_RES);
    }

    // ---- lm_head with fused slice/scatter into (B, CODEC, 513, VOCAB) ----
    k_gemm<<<dim3((CODEC * VOCAB) / 64, ROWS / 128), gB, 0, stream>>>(
        h, lm_head, (float*)d_out, nullptr, nullptr,
        ROWS, CODEC * VOCAB, D_MODEL, EP_LMHEAD);
}
